// EquiCategorySpecificMLP_11184094839061
// MI455X (gfx1250) — matrix operand order repaired and checked by reading, not compiled, not run on hardware
//
#include <hip/hip_runtime.h>
#include <hip/hip_bf16.h>

typedef __bf16 v16bf __attribute__((ext_vector_type(16)));
typedef float  v8f   __attribute__((ext_vector_type(8)));
typedef unsigned short u16;

#define C_CATS 16
#define D_IN   256
#define D_H    1024
#define D_OUT  256
#define M_TILE 64
#define BLOCK  256

// LDS layout (dynamic): X tile (64x256 bf16) | H tile (64x1024 bf16) | perm tile (64 int)
#define SX_BYTES   (M_TILE * D_IN * 2)            // 32768
#define SH_BYTES   (M_TILE * D_H * 2)             // 131072
#define SPERM_BYTES (M_TILE * 4)                  // 256
#define SMEM_BYTES (SX_BYTES + SH_BYTES + SPERM_BYTES)

union FragBF {
    v16bf v;
    uint4 q[2];
};

__device__ __forceinline__ u16 f2bf(float f) {
    union { __bf16 b; u16 u; } cv;
    cv.b = (__bf16)f;
    return cv.u;
}

// ---------------- prep kernels ----------------

__global__ void zero_kernel(int* counts, int* cursor) {
    int t = threadIdx.x;
    if (t < C_CATS) { counts[t] = 0; cursor[t] = 0; }
}

__global__ void hist_kernel(const int* __restrict__ cat, int B, int* __restrict__ counts) {
    __shared__ int bins[C_CATS];
    int t = threadIdx.x;
    if (t < C_CATS) bins[t] = 0;
    __syncthreads();
    int i = blockIdx.x * blockDim.x + t;
    if (i < B) atomicAdd(&bins[cat[i]], 1);
    __syncthreads();
    if (t < C_CATS && bins[t] != 0) atomicAdd(&counts[t], bins[t]);
}

__global__ void scan_tiles_kernel(const int* __restrict__ counts, int* __restrict__ cursor,
                                  int4* __restrict__ tiles, int maxT) {
    int t = threadIdx.x;
    for (int i = t; i < maxT; i += blockDim.x) tiles[i] = make_int4(0, 0, 0, 0);
    __syncthreads();
    if (t == 0) {
        int off = 0, tid = 0;
        for (int c = 0; c < C_CATS; ++c) {
            int n = counts[c];
            cursor[c] = off;
            for (int s = 0; s < n; s += M_TILE) {
                int len = n - s; if (len > M_TILE) len = M_TILE;
                tiles[tid++] = make_int4(c, off + s, len, 0);
            }
            off += n;
        }
    }
}

__global__ void scatter_kernel(const int* __restrict__ cat, int B,
                               int* __restrict__ cursor, int* __restrict__ perm) {
    int i = blockIdx.x * blockDim.x + threadIdx.x;
    if (i < B) {
        int p = atomicAdd(&cursor[cat[i]], 1);
        perm[p] = i;
    }
}

// Transpose + convert weights to bf16: W1t[c][n][k] (n<D_H,k<D_IN), W2t[c][n][k] (n<D_OUT,k<D_H)
__global__ void convert_w_kernel(const float* __restrict__ W1, const float* __restrict__ W2,
                                 u16* __restrict__ W1t, u16* __restrict__ W2t) {
    const long total = (long)C_CATS * D_H * D_IN;  // same element count for both matrices
    long i = (long)blockIdx.x * blockDim.x + threadIdx.x;
    if (i < total) {
        int k = (int)(i % D_IN);
        int n = (int)((i / D_IN) % D_H);
        int c = (int)(i / ((long)D_IN * D_H));
        W1t[i] = f2bf(W1[((size_t)c * D_IN + k) * D_H + n]);
    } else {
        long j = i - total;
        int k = (int)(j % D_H);
        int n = (int)((j / D_H) % D_OUT);
        int c = (int)(j / ((long)D_H * D_OUT));
        W2t[j] = f2bf(W2[((size_t)c * D_H + k) * D_OUT + n]);
    }
}

// ---------------- fused routed-MLP kernel ----------------

__global__ __launch_bounds__(BLOCK, 2) void mlp_kernel(
    const float* __restrict__ x,
    const float* __restrict__ b1,
    const float* __restrict__ b2,
    const u16*  __restrict__ W1t,
    const u16*  __restrict__ W2t,
    const int*  __restrict__ perm,
    const int4* __restrict__ tiles,
    float* __restrict__ out)
{
    extern __shared__ char smem_raw[];
    u16* sX    = (u16*)smem_raw;                           // [64][256] bf16
    u16* sH    = (u16*)(smem_raw + SX_BYTES);              // [64][1024] bf16
    int* sPerm = (int*)(smem_raw + SX_BYTES + SH_BYTES);   // [64]

    int4 td = tiles[blockIdx.x];
    const int cat = td.x, start = td.y, cnt = td.z;
    if (cnt <= 0) return;

    const int t    = threadIdx.x;
    const int wave = t >> 5;
    const int lane = t & 31;
    const int nrow = lane & 15;   // N column (B/C/D frags) or M row (A frags)
    const int half = lane >> 4;

    if (t < M_TILE) sPerm[t] = (t < cnt) ? perm[start + t] : 0;
    __syncthreads();

    // Gather x tile -> LDS bf16 (zero-pad rows >= cnt). 64 rows x 64 float4.
    for (int i = t; i < M_TILE * (D_IN / 4); i += BLOCK) {
        int row = i >> 6;
        int c4  = i & 63;
        float4 f = make_float4(0.f, 0.f, 0.f, 0.f);
        if (row < cnt) {
            const float4* src = (const float4*)(x + (size_t)sPerm[row] * D_IN);
            f = src[c4];
        }
        ushort4 h;
        h.x = f2bf(f.x); h.y = f2bf(f.y); h.z = f2bf(f.z); h.w = f2bf(f.w);
        *(ushort4*)(sX + row * D_IN + c4 * 4) = h;
    }
    __syncthreads();

    // ---- Layer 1: H = relu(X @ W1[cat] + b1[cat]); M=64 K=256 N=1024 ----
    // Each wave owns 128 N-columns (8 n-tiles), loops 4 m-tiles, K=8 wmma steps.
    for (int nt = 0; nt < 8; ++nt) {
        const int n = wave * 128 + nt * 16 + nrow;
        const u16* wrow = W1t + ((size_t)(cat * D_H + n)) * D_IN;
        FragBF Bf[8];
#pragma unroll
        for (int kk = 0; kk < 8; ++kk) {
            int kb = kk * 32 + (half << 3);               // B layout: K 0-7/16-23 vs 8-15/24-31
            Bf[kk].q[0] = *(const uint4*)(wrow + kb);
            Bf[kk].q[1] = *(const uint4*)(wrow + kb + 16);
        }
        const float bias = b1[cat * D_H + n];
#pragma unroll
        for (int mt = 0; mt < 4; ++mt) {
            v8f acc;
#pragma unroll
            for (int v = 0; v < 8; ++v) acc[v] = bias;
#pragma unroll
            for (int kk = 0; kk < 8; ++kk) {
                FragBF Af;
                int r  = mt * 16 + nrow;
                int ka = kk * 32 + (half << 3);           // A layout: K 0-7/16-23 vs 8-15/24-31
                const u16* xr = sX + r * D_IN;
                Af.q[0] = *(const uint4*)(xr + ka);
                Af.q[1] = *(const uint4*)(xr + ka + 16);
                acc = __builtin_amdgcn_wmma_f32_16x16x32_bf16(
                    false, Af.v, false, Bf[kk].v, (short)0, acc, false, false);
            }
#pragma unroll
            for (int v = 0; v < 8; ++v) {
                int m = mt * 16 + (half << 3) + v;        // C layout: M = vgpr + 8*half
                sH[m * D_H + n] = f2bf(fmaxf(acc[v], 0.0f));
            }
        }
    }
    __syncthreads();

    // ---- Layer 2: OUT = H @ W2[cat] + b2[cat]; M=64 K=1024 N=256 ----
    // Each wave owns 32 N-columns (2 n-tiles), loops 4 m-tiles, K=32 wmma steps.
    for (int nt = 0; nt < 2; ++nt) {
        const int n = wave * 32 + nt * 16 + nrow;
        const u16* wrow = W2t + ((size_t)(cat * D_OUT + n)) * D_H;
        const float bias = b2[cat * D_OUT + n];
#pragma unroll
        for (int mt = 0; mt < 4; ++mt) {
            v8f acc;
#pragma unroll
            for (int v = 0; v < 8; ++v) acc[v] = bias;
#pragma unroll 4
            for (int kk = 0; kk < 32; ++kk) {
                FragBF Af, Bf;
                int r  = mt * 16 + nrow;
                int ka = kk * 32 + (half << 3);
                const u16* hr = sH + r * D_H;
                Af.q[0] = *(const uint4*)(hr + ka);
                Af.q[1] = *(const uint4*)(hr + ka + 16);
                int kb = kk * 32 + (half << 3);
                Bf.q[0] = *(const uint4*)(wrow + kb);
                Bf.q[1] = *(const uint4*)(wrow + kb + 16);
                acc = __builtin_amdgcn_wmma_f32_16x16x32_bf16(
                    false, Af.v, false, Bf.v, (short)0, acc, false, false);
            }
#pragma unroll
            for (int v = 0; v < 8; ++v) {
                int m = mt * 16 + (half << 3) + v;
                if (m < cnt) out[(size_t)sPerm[m] * D_OUT + n] = acc[v];
            }
        }
    }
}

// ---------------- launcher ----------------

extern "C" void kernel_launch(void* const* d_in, const int* in_sizes, int n_in,
                              void* d_out, int out_size, void* d_ws, size_t ws_size,
                              hipStream_t stream) {
    (void)n_in; (void)out_size; (void)ws_size;
    const float* x   = (const float*)d_in[0];
    const int*   cat = (const int*)d_in[1];
    const float* W1  = (const float*)d_in[2];
    const float* b1  = (const float*)d_in[3];
    const float* W2  = (const float*)d_in[4];
    const float* b2  = (const float*)d_in[5];
    float* out = (float*)d_out;

    const int B    = in_sizes[1];
    const int maxT = (B + M_TILE - 1) / M_TILE + C_CATS;  // upper bound on category tiles

    // workspace carve-out
    char* ws = (char*)d_ws;
    size_t off = 0;
    auto take = [&](size_t bytes) -> char* {
        char* p = ws + off;
        off = (off + bytes + 255) & ~(size_t)255;
        return p;
    };
    u16*  W1t    = (u16*)take((size_t)C_CATS * D_H * D_IN * 2);
    u16*  W2t    = (u16*)take((size_t)C_CATS * D_OUT * D_H * 2);
    int*  perm   = (int*)take((size_t)B * 4);
    int*  counts = (int*)take(C_CATS * 4);
    int*  cursor = (int*)take(C_CATS * 4);
    int4* tiles  = (int4*)take((size_t)maxT * sizeof(int4));

    hipLaunchKernelGGL(zero_kernel, dim3(1), dim3(32), 0, stream, counts, cursor);
    hipLaunchKernelGGL(hist_kernel, dim3((B + 255) / 256), dim3(256), 0, stream, cat, B, counts);
    hipLaunchKernelGGL(scan_tiles_kernel, dim3(1), dim3(256), 0, stream, counts, cursor, tiles, maxT);
    hipLaunchKernelGGL(scatter_kernel, dim3((B + 255) / 256), dim3(256), 0, stream, cat, B, cursor, perm);

    const long totalW = 2L * C_CATS * D_H * D_IN;         // both matrices, same element count
    hipLaunchKernelGGL(convert_w_kernel, dim3((unsigned)(totalW / 256)), dim3(256), 0, stream,
                       W1, W2, W1t, W2t);

    hipFuncSetAttribute((const void*)mlp_kernel,
                        hipFuncAttributeMaxDynamicSharedMemorySize, (int)SMEM_BYTES);
    hipLaunchKernelGGL(mlp_kernel, dim3(maxT), dim3(BLOCK), SMEM_BYTES, stream,
                       x, b1, b2, W1t, W2t, perm, tiles, out);
}
